// TopKSAE_54537494725080
// MI455X (gfx1250) — matrix operand order repaired and checked by reading, not compiled, not run on hardware
//
#include <hip/hip_runtime.h>

typedef __bf16 v16bf __attribute__((ext_vector_type(16)));
typedef float  v8f   __attribute__((ext_vector_type(8)));
typedef unsigned short u16;
typedef unsigned int   u32;

#define N_TOK 4096
#define D_IN  2048
#define D_SAE 32768
#define TOPK  64

#define ROWS   16                 // token rows per workgroup (one WMMA M-tile)
#define WAVES  8                  // waves per workgroup
#define STRIP  8                  // C tiles held in registers per wave (8 * v8f = 64 VGPRs)
#define KT     (D_IN / 32)        // 64 k-tiles of 32 (bf16 WMMA K)
#define NT_WAVE (D_SAE / 16 / WAVES)  // 256 column tiles per wave

__device__ __forceinline__ u16 f2bf(float f) {      // fp32 -> bf16 round-to-nearest-even
  u32 u = __float_as_uint(f);
  u += 0x7fffu + ((u >> 16) & 1u);
  return (u16)(u >> 16);
}

// ---------------------------------------------------------------------------
// Kernel 1: convert W_enc [2048 x 32768] fp32 (row-major, k-major rows) into
// bf16, pre-swizzled into the WMMA B-operand layout:
//   wsb[(ntile*KT + kt)*32 + lane][16 bf16]  where lane n<16 holds K=kt*32+0..15
//   at N=ntile*16+n, lanes 16..31 hold K=kt*32+16..31. One contiguous 32B load
//   per lane per WMMA in the GEMM.
// ---------------------------------------------------------------------------
__global__ __launch_bounds__(256) void wenc_pack(const float* __restrict__ W,
                                                 u16* __restrict__ wsb) {
  __shared__ u16 t[32][264];                       // 32k x 256n tile, padded
  const int nb = blockIdx.x * 256;                 // n base
  const int kb = blockIdx.y * 32;                  // k base (= kt*32)
  const int tid = threadIdx.x;
  #pragma unroll 4
  for (int k = 0; k < 32; ++k)
    t[k][tid] = f2bf(W[(size_t)(kb + k) * D_SAE + nb + tid]);
  __syncthreads();
  // 16 ntiles * 32 lanes = 512 output slots of 32B, 2 per thread
  for (int s = tid; s < 512; s += 256) {
    const int nt = s >> 5, l = s & 31;
    const int nl = nt * 16 + (l & 15);
    const int k0 = (l >> 4) ? 16 : 0;
    uint4 lo, hi;
    lo.x = (u32)t[k0 + 0][nl] | ((u32)t[k0 + 1][nl] << 16);
    lo.y = (u32)t[k0 + 2][nl] | ((u32)t[k0 + 3][nl] << 16);
    lo.z = (u32)t[k0 + 4][nl] | ((u32)t[k0 + 5][nl] << 16);
    lo.w = (u32)t[k0 + 6][nl] | ((u32)t[k0 + 7][nl] << 16);
    hi.x = (u32)t[k0 + 8][nl] | ((u32)t[k0 + 9][nl] << 16);
    hi.y = (u32)t[k0 +10][nl] | ((u32)t[k0 +11][nl] << 16);
    hi.z = (u32)t[k0 +12][nl] | ((u32)t[k0 +13][nl] << 16);
    hi.w = (u32)t[k0 +14][nl] | ((u32)t[k0 +15][nl] << 16);
    u16* dst = wsb + ((((size_t)(nb / 16 + nt)) * KT + blockIdx.y) * 32 + l) * 16;
    *(uint4*)dst = lo;
    *((uint4*)dst + 1) = hi;
  }
}

// ---------------------------------------------------------------------------
// Kernel 2: W_dec fp32 -> bf16 (row-major, unchanged layout). 128MB fits L2,
// so the sparse-decode row gather mostly hits L2.
// ---------------------------------------------------------------------------
__global__ __launch_bounds__(256) void wdec_conv(const float* __restrict__ W,
                                                 u32* __restrict__ dst) {
  const size_t t = (size_t)blockIdx.x * 256 + threadIdx.x;  // one u32 = 2 bf16
  const size_t i = t * 2;
  dst[t] = (u32)f2bf(W[i]) | ((u32)f2bf(W[i + 1]) << 16);
}

// ---------------------------------------------------------------------------
// Kernel 3: A' = acts - b_dec, fp32 -> bf16, row-major [4096][2048]
// ---------------------------------------------------------------------------
__global__ __launch_bounds__(256) void acts_pack(const float* __restrict__ acts,
                                                 const float* __restrict__ b_dec,
                                                 u32* __restrict__ dst) {
  const size_t t = (size_t)blockIdx.x * 256 + threadIdx.x;
  const size_t i = t * 2;
  const int k = (int)(i & (D_IN - 1));
  const u16 a = f2bf(acts[i] - b_dec[k]);
  const u16 b = f2bf(acts[i + 1] - b_dec[k + 1]);
  dst[t] = (u32)a | ((u32)b << 16);
}

// ---------------------------------------------------------------------------
// Kernel 4: fused encode-GEMM (bf16 WMMA 16x16x32) + ReLU + exact top-64 +
// sparse gather decode. One workgroup = 16 token rows, 8 waves; wave w owns
// columns [w*4096, (w+1)*4096).
// ---------------------------------------------------------------------------
__global__ __launch_bounds__(256) void sae_fused(const u16* __restrict__ wsb,
                                                 const u16* __restrict__ wsa,
                                                 const u16* __restrict__ wsd,
                                                 const float* __restrict__ b_enc,
                                                 const float* __restrict__ b_dec,
                                                 float* __restrict__ out) {
  __shared__ u16   a_sw[KT * 32 * 16];          // 64KB: A' swizzled for WMMA A operand
  __shared__ float hval[WAVES][ROWS][TOPK];     // 32KB: per-wave min-heaps (values)
  __shared__ int   hidx[WAVES][ROWS][TOPK];     // 32KB: per-wave min-heaps (indices)
  __shared__ float ctile[WAVES][ROWS][17];      // 8.5KB: C-tile staging (padded)
  __shared__ float fval[ROWS][TOPK];            // 4KB: merged top-64 values
  __shared__ int   fidx[ROWS][TOPK];            // 4KB: merged top-64 indices

  const int tid  = threadIdx.x;
  const int wave = tid >> 5;
  const int lane = tid & 31;
  const int row0 = blockIdx.x * ROWS;

  // ---- stage A' (16 rows x 2048 K, bf16) into LDS in WMMA-A lane layout ----
  // slot (kt, l): lane l (m=l&15, hi=l>>4) reads elems 0..7 = K(kt*32 + hi?16:0 ..)
  // and elems 8..15 = K(kt*32 + hi?24:8 ..), contiguous K order matching packed B.
  for (int s = tid; s < KT * 32; s += 256) {
    const int kt = s >> 5, l = s & 31, m = l & 15, hi = l >> 4;
    const u16* src = wsa + (size_t)(row0 + m) * D_IN + kt * 32;
    u16* dst = &a_sw[s * 16];
    *(uint4*)dst       = *(const uint4*)(src + (hi ? 16 : 0));
    *(uint4*)(dst + 8) = *(const uint4*)(src + (hi ? 24 : 8));
  }
  // ---- init heaps: min value 0 => ReLU'd non-positive pre-acts never enter ----
  for (int s = tid; s < WAVES * ROWS * TOPK; s += 256) {
    (&hval[0][0][0])[s] = 0.f;
    (&hidx[0][0][0])[s] = 0;
  }
  __syncthreads();

  const v8f vzero = {0.f, 0.f, 0.f, 0.f, 0.f, 0.f, 0.f, 0.f};
  const int m  = lane & 15;
  const int hi = lane >> 4;

  // ---- encode GEMM + streaming per-row top-64 ----
  for (int strip = 0; strip < NT_WAVE / STRIP; ++strip) {
    const int nt0 = wave * NT_WAVE + strip * STRIP;   // global ntile base
    v8f acc[STRIP];
    #pragma unroll
    for (int t = 0; t < STRIP; ++t) acc[t] = vzero;

    for (int kt = 0; kt < KT; ++kt) {
      const v16bf a = *(const v16bf*)&a_sw[(kt * 32 + lane) * 16];
      #pragma unroll
      for (int t = 0; t < STRIP; ++t) {
        const v16bf b = *(const v16bf*)(wsb +
            ((((size_t)(nt0 + t)) * KT + kt) * 32 + lane) * 16);
        acc[t] = __builtin_amdgcn_wmma_f32_16x16x32_bf16(
            false, a, false, b, (short)0, acc[t], false, false);
      }
    }

    // epilogue: +b_enc, ReLU, funnel candidates to row-owner lanes via LDS heap
    for (int t = 0; t < STRIP; ++t) {
      const int coltile = (nt0 + t) * 16;
      const float be = b_enc[coltile + m];     // C layout: N = lane&15
      #pragma unroll
      for (int r = 0; r < 8; ++r)              // C layout: M = r + 8*(lane>>4)
        ctile[wave][r + hi * 8][m] = fmaxf(acc[t][r] + be, 0.f);

      if (lane < 16) {                         // lane owns row m = lane
        float* hv = &hval[wave][lane][0];
        int*   hx = &hidx[wave][lane][0];
        for (int cc = 0; cc < 16; ++cc) {
          const float v = ctile[wave][lane][cc];
          if (v > hv[0]) {                     // beats current 64th-largest
            int p = 0;                         // sift-down in 64-entry min-heap
            for (;;) {
              const int l2 = 2 * p + 1, r2 = 2 * p + 2;
              int s2 = p; float sv = v;
              if (l2 < TOPK && hv[l2] < sv) { s2 = l2; sv = hv[l2]; }
              if (r2 < TOPK && hv[r2] < sv) { s2 = r2; }
              if (s2 == p) break;
              hv[p] = hv[s2]; hx[p] = hx[s2];
              p = s2;
            }
            hv[p] = v; hx[p] = coltile + cc;
          }
        }
      }
    }
  }
  __syncthreads();

  // ---- merge 8 partial top-64 lists per row (wave handles rows 2w, 2w+1) ----
  for (int rr = 0; rr < 2; ++rr) {
    const int mr = wave * 2 + rr;
    for (int j = 0; j < TOPK; ++j) {
      float bv = -1.f; int bc = 0;
      #pragma unroll
      for (int c0 = 0; c0 < 16; ++c0) {        // 512 candidates / 32 lanes
        const int c = lane + c0 * 32;
        const float v = hval[c >> 6][mr][c & 63];
        if (v > bv) { bv = v; bc = c; }
      }
      #pragma unroll
      for (int off = 16; off; off >>= 1) {     // butterfly max (val, candidate)
        const float ov = __shfl_xor(bv, off, 32);
        const int   oc = __shfl_xor(bc, off, 32);
        if (ov > bv || (ov == bv && oc < bc)) { bv = ov; bc = oc; }
      }
      if (lane == 0) {
        fval[mr][j] = bv > 0.f ? bv : 0.f;
        fidx[mr][j] = bv > 0.f ? hidx[bc >> 6][mr][bc & 63] : 0;
      }
      if (lane == (bc & 31))                   // owner lane retires the winner
        hval[bc >> 6][mr][bc & 63] = -1.f;
    }
  }
  __syncthreads();

  // ---- sparse decode: recon[row] = b_dec + sum_j fval * W_dec_bf16[fidx] ----
  const int col0 = wave * 256 + lane * 8;      // 8 contiguous cols per lane
  for (int r = 0; r < ROWS; ++r) {
    const float4 bd0 = *(const float4*)(b_dec + col0);
    const float4 bd1 = *(const float4*)(b_dec + col0 + 4);
    float a0 = bd0.x, a1 = bd0.y, a2 = bd0.z, a3 = bd0.w;
    float a4 = bd1.x, a5 = bd1.y, a6 = bd1.z, a7 = bd1.w;
    for (int j = 0; j < TOPK; ++j) {
      const float v = fval[r][j];              // LDS broadcast (wave-uniform)
      const int idx = fidx[r][j];
      const uint4 raw = *(const uint4*)(wsd + (size_t)idx * D_IN + col0);
      a0 += v * __uint_as_float(raw.x << 16);
      a1 += v * __uint_as_float(raw.x & 0xffff0000u);
      a2 += v * __uint_as_float(raw.y << 16);
      a3 += v * __uint_as_float(raw.y & 0xffff0000u);
      a4 += v * __uint_as_float(raw.z << 16);
      a5 += v * __uint_as_float(raw.z & 0xffff0000u);
      a6 += v * __uint_as_float(raw.w << 16);
      a7 += v * __uint_as_float(raw.w & 0xffff0000u);
    }
    float4 o0 = {a0, a1, a2, a3}, o1 = {a4, a5, a6, a7};
    float4* op = (float4*)(out + (size_t)(row0 + r) * D_IN + col0);
    op[0] = o0; op[1] = o1;
  }
}

// ---------------------------------------------------------------------------
// Host-side launch. Workspace layout (assumes ws_size >= 272 MB):
//   [0,   128MB) W_enc swizzled bf16    [128MB, 256MB) W_dec bf16
//   [256MB, 272MB) A' = acts - b_dec, bf16
// ---------------------------------------------------------------------------
extern "C" void kernel_launch(void* const* d_in, const int* in_sizes, int n_in,
                              void* d_out, int out_size, void* d_ws, size_t ws_size,
                              hipStream_t stream) {
  const float* acts  = (const float*)d_in[0];
  const float* W_enc = (const float*)d_in[1];
  const float* W_dec = (const float*)d_in[2];
  const float* b_enc = (const float*)d_in[3];
  const float* b_dec = (const float*)d_in[4];
  float* out = (float*)d_out;

  char* ws = (char*)d_ws;
  u16* wsb = (u16*)ws;
  u16* wsd = (u16*)(ws + (size_t)128 * 1024 * 1024);
  u16* wsa = (u16*)(ws + (size_t)256 * 1024 * 1024);

  dim3 g1(D_SAE / 256, KT);
  wenc_pack<<<g1, 256, 0, stream>>>(W_enc, wsb);

  wdec_conv<<<((size_t)D_SAE * D_IN / 2) / 256, 256, 0, stream>>>(W_dec, (u32*)wsd);

  acts_pack<<<((size_t)N_TOK * D_IN / 2) / 256, 256, 0, stream>>>(acts, b_dec, (u32*)wsa);

  sae_fused<<<N_TOK / ROWS, 256, 0, stream>>>(wsb, wsa, wsd, b_enc, b_dec, out);
}
